// NeuroSAT_39264591020540
// MI455X (gfx1250) — matrix operand order repaired and checked by reading, not compiled, not run on hardware
//
#include <hip/hip_runtime.h>
#include <cmath>

#define EMB 128
static constexpr int V_  = 50000;
static constexpr int C_  = 200000;
static constexpr int E_  = 300000;
static constexpr int NL_ = 2 * V_;      // literals (pos+neg)
static constexpr int ROUNDS_ = 26;

typedef __attribute__((ext_vector_type(16))) _Float16 v16h;
typedef __attribute__((ext_vector_type(8)))  _Float16 v8h;
typedef __attribute__((ext_vector_type(4)))  _Float16 v4h;
typedef __attribute__((ext_vector_type(8)))  float    v8f;

// ---------------- WMMA helpers (ISA 7.12.2 layouts, wave32) ----------------

__device__ __forceinline__ v8f zero8() {
  v8f z;
#pragma unroll
  for (int i = 0; i < 8; ++i) z[i] = 0.f;
  return z;
}

// A: 16x32 f16 tile, row-major, ld must be a multiple of 8 halfs.
// lanes 0-15: row=lane, halfs 0-15 -> K kb..kb+15 (kb=0)
// lanes 16-31: same rows, kb=16. One 32-byte contiguous run -> 2x ds_load_b128.
__device__ __forceinline__ v16h frag_A(const _Float16* __restrict__ x, int ld, int lane) {
  int row = lane & 15;
  int kb  = (lane & 16) ? 16 : 0;
  const v8h* p = (const v8h*)(x + row * ld + kb);
  v8h lo = p[0];
  v8h hi = p[1];  // +8 halfs
  return __builtin_shufflevector(lo, hi, 0, 1, 2, 3, 4, 5, 6, 7,
                                 8, 9, 10, 11, 12, 13, 14, 15);
}

// B fragment from pre-swizzled tile storage: [lane:32][j:16] halfs, contiguous per lane.
__device__ __forceinline__ v16h frag_B_sw(const _Float16* __restrict__ tile, int lane) {
  const v8h* p = (const v8h*)(tile + (lane << 4));
  v8h lo = p[0];
  v8h hi = p[1];
  return __builtin_shufflevector(lo, hi, 0, 1, 2, 3, 4, 5, 6, 7,
                                 8, 9, 10, 11, 12, 13, 14, 15);
}

__device__ __forceinline__ v8f wmma16(v16h a, v16h b, v8f c) {
  return __builtin_amdgcn_wmma_f32_16x16x32_f16(false, a, false, b, (short)0, c,
                                                false, false);
}

__device__ __forceinline__ float sigmoidf_(float x) { return 1.f / (1.f + __expf(-x)); }

// ---------------- weight preparation kernels ----------------
// Swizzle W (K x N row-major f32) into f16 fragment-native order:
// out[[kc:K/32][nt:N/16][lane:32][j:16]] = W[kc*32 + ((lane&16)?16:0) + j][nt*16 + (lane&15)]

__global__ void swz_b_kernel(const float* __restrict__ W, _Float16* __restrict__ out,
                             int K, int N) {
  int idx = blockIdx.x * 256 + threadIdx.x;
  if (idx >= K * N) return;
  int tile = idx >> 9;          // 512 halfs per 32x16 tile
  int lane = (idx >> 4) & 31;
  int j    = idx & 15;
  int ntiles = N >> 4;
  int kc = tile / ntiles, nt = tile % ntiles;
  int n = nt * 16 + (lane & 15);
  int k = kc * 32 + ((lane & 16) ? 16 : 0) + j;
  out[idx] = (_Float16)W[k * N + n];
}

// Swizzled concatenated LSTM weight: logical Wcat[(KX+128) x 512],
// rows 0..KX-1 = Wih.T (Wih is 512 x KX), rows KX.. = Whh.T (Whh is 512 x 128).
__global__ void build_wcat_swz_kernel(const float* __restrict__ Wih,
                                      const float* __restrict__ Whh,
                                      _Float16* __restrict__ out, int KX) {
  int KT = KX + 128;
  int idx = blockIdx.x * 256 + threadIdx.x;
  if (idx >= KT * 512) return;
  int tile = idx >> 9;
  int lane = (idx >> 4) & 31;
  int j    = idx & 15;
  int ntiles = 512 >> 4;  // 32
  int kc = tile / ntiles, nt = tile % ntiles;
  int n = nt * 16 + (lane & 15);
  int k = kc * 32 + ((lane & 16) ? 16 : 0) + j;
  float v = (k < KX) ? Wih[n * KX + k] : Whh[n * 128 + (k - KX)];
  out[idx] = (_Float16)v;
}

__global__ void add_bias_kernel(const float* __restrict__ a, const float* __restrict__ b,
                                float* __restrict__ o, int n) {
  int i = blockIdx.x * 256 + threadIdx.x;
  if (i < n) o[i] = a[i] + b[i];
}

// ---------------- 3-layer MLP: Y = L3(relu(L2(relu(L1(X))))) ----------------
// 16 rows/block, 8 waves; wave w -> output cols [16w, 16w+16).
// W16sw: per layer [4][8][512] swizzled halfs (16384/layer).

static constexpr int LDX_MLP = 136;  // padded ld (halfs), multiple of 8, not pow2

__global__ __launch_bounds__(256) void mlp3_kernel(
    const float* __restrict__ X, const _Float16* __restrict__ W16sw,
    const float* __restrict__ Bias, float* __restrict__ Y, int N) {
  __shared__ _Float16 sX[16 * LDX_MLP];
  __shared__ __align__(16) _Float16 sW[16384];

  int t = threadIdx.x, lane = t & 31, wave = t >> 5;
  int row0 = blockIdx.x * 16;

  // stage X tile (float4 global loads -> v4h LDS stores)
  for (int i = t; i < 16 * 32; i += 256) {
    int r = i >> 5, c4 = i & 31;
    int gr = row0 + r;
    if (gr >= N) gr = N - 1;
    float4 v = ((const float4*)(X + (size_t)gr * EMB))[c4];
    v4h h = {(_Float16)v.x, (_Float16)v.y, (_Float16)v.z, (_Float16)v.w};
    *(v4h*)(&sX[r * LDX_MLP + c4 * 4]) = h;
  }

  for (int layer = 0; layer < 3; ++layer) {
    __syncthreads();
    {  // linear 32 KB copy of the swizzled layer (uint4 = 8 halfs)
      const uint4* src = (const uint4*)(W16sw + layer * 16384);
      uint4* dst = (uint4*)sW;
      for (int i = t; i < 2048; i += 256) dst[i] = src[i];
    }
    __syncthreads();

    v8f acc = zero8();
#pragma unroll
    for (int kc = 0; kc < 4; ++kc) {
      v16h a = frag_A(sX + kc * 32, LDX_MLP, lane);
      v16h b = frag_B_sw(sW + (kc * 8 + wave) * 512, lane);
      acc = wmma16(a, b, acc);
    }
    __syncthreads();  // all reads of sX/sW complete

    const float* bias = Bias + layer * EMB;
    int cidx = wave * 16 + (lane & 15);
    float bb = bias[cidx];
#pragma unroll
    for (int i = 0; i < 8; ++i) {
      int r = (lane & 16) ? (i + 8) : i;
      float v = acc[i] + bb;
      if (layer < 2) {
        sX[r * LDX_MLP + cidx] = (_Float16)(v > 0.f ? v : 0.f);
      } else {
        int gr = row0 + r;
        if (gr < N) Y[gr * EMB + cidx] = v;
      }
    }
  }
}

// ---------------- fused LSTM cell ----------------
// gates = [x|h] @ Wcat + bsum ; then elementwise. 16 rows/block, 8 waves;
// wave w -> gate cols [64w, 64w+64). KX = x width (256 literal / 128 clause).
// Wcatsw: [(KX+128)/32][32][32][16] swizzled halfs; 16384 per K-chunk.

template <int KX>
__global__ __launch_bounds__(256) void lstm_kernel(
    const float* __restrict__ Xa,       // N x 128 (c2l or l2c)
    const float* __restrict__ Hin,      // N x 128
    float* __restrict__ Cst,            // N x 128 (in/out)
    float* __restrict__ Hout,           // N x 128
    const _Float16* __restrict__ Wcatsw,
    const float* __restrict__ bsum,     // 512
    int N, int Vhalf) {
  constexpr int KT  = KX + 128;
  constexpr int LDX = KT + 8;  // padded ld, multiple of 8
  __shared__ _Float16 sX[16 * LDX];
  __shared__ __align__(16) char sBuf[32 * 512 * 2];  // f16 W-chunk, aliased by f32 gates
  _Float16* sW = (_Float16*)sBuf;
  float*    sG = (float*)sBuf;

  int t = threadIdx.x, lane = t & 31, wave = t >> 5;
  int row0 = blockIdx.x * 16;

  // stage [x | (flip) | h] as f16, float4 granularity (source regions 128-aligned)
  constexpr int C4 = KT / 4;
  for (int i = t; i < 16 * C4; i += 256) {
    int r = i / C4, c = (i % C4) * 4;
    int gr = row0 + r;
    if (gr >= N) gr = N - 1;
    const float* src;
    if (c < 128) {
      src = Xa + (size_t)gr * 128 + c;
    } else if (KX == 256 && c < 256) {
      int fr = (gr < Vhalf) ? (gr + Vhalf) : (gr - Vhalf);
      src = Hin + (size_t)fr * 128 + (c - 128);
    } else {
      src = Hin + (size_t)gr * 128 + (c - (KT - 128));
    }
    float4 v = *(const float4*)src;
    v4h h = {(_Float16)v.x, (_Float16)v.y, (_Float16)v.z, (_Float16)v.w};
    *(v4h*)(&sX[r * LDX + c]) = h;
  }

  v8f acc[4];
#pragma unroll
  for (int ct = 0; ct < 4; ++ct) acc[ct] = zero8();

#pragma unroll
  for (int kc = 0; kc < KT / 32; ++kc) {
    __syncthreads();  // prior frag_B reads done; sX visible on first iter
    {
      const uint4* src = (const uint4*)(Wcatsw + kc * 16384);
      uint4* dst = (uint4*)sW;
      for (int i = t; i < 2048; i += 256) dst[i] = src[i];
    }
    if (kc + 1 < KT / 32)  // warm L2/L0 for next chunk (global_prefetch)
      __builtin_prefetch(Wcatsw + (kc + 1) * 16384 + t * 64, 0, 0);
    __syncthreads();
    v16h a = frag_A(sX + kc * 32, LDX, lane);
#pragma unroll
    for (int ct = 0; ct < 4; ++ct) {
      v16h b = frag_B_sw(sW + (wave * 4 + ct) * 512, lane);
      acc[ct] = wmma16(a, b, acc[ct]);
    }
  }
  __syncthreads();  // done with sW -> safe to alias as gates

#pragma unroll
  for (int ct = 0; ct < 4; ++ct) {
    int cidx = wave * 64 + ct * 16 + (lane & 15);
    float bb = bsum[cidx];
#pragma unroll
    for (int i = 0; i < 8; ++i) {
      int r = (lane & 16) ? (i + 8) : i;
      sG[r * 512 + cidx] = acc[ct][i] + bb;
    }
  }
  __syncthreads();

  for (int e = t; e < 16 * 128; e += 256) {
    int r = e >> 7, j = e & 127;
    int gr = row0 + r;
    if (gr < N) {
      float gi = sG[r * 512 + j];
      float gf = sG[r * 512 + 128 + j];
      float gg = sG[r * 512 + 256 + j];
      float go = sG[r * 512 + 384 + j];
      float co = Cst[gr * 128 + j];
      float cn = sigmoidf_(gf) * co + sigmoidf_(gi) * tanhf(gg);
      Cst[gr * 128 + j]  = cn;
      Hout[gr * 128 + j] = sigmoidf_(go) * tanhf(cn);
    }
  }
}

// ---------------- edge scatter: Out[dst] += Msg[src], one wave per edge ----------------

__global__ __launch_bounds__(256) void scatter_add_kernel(
    const float* __restrict__ Msg, const int* __restrict__ srcIdx,
    const int* __restrict__ dstIdx, float* __restrict__ Out,
    int E, int srcBase, int dstBase) {
  int gw = (int)((blockIdx.x * 256u + threadIdx.x) >> 5);
  int lane = threadIdx.x & 31;
  if (gw >= E) return;
  int s = srcIdx[gw] + srcBase;
  int d = dstIdx[gw] + dstBase;
  float4 v = ((const float4*)(Msg + (size_t)s * 128))[lane];
  float* out = Out + (size_t)d * 128 + lane * 4;
  atomicAdd(out + 0, v.x);
  atomicAdd(out + 1, v.y);
  atomicAdd(out + 2, v.z);
  atomicAdd(out + 3, v.w);
}

// ---------------- host orchestration ----------------

extern "C" void kernel_launch(void* const* d_in, const int* in_sizes, int n_in,
                              void* d_out, int out_size, void* d_ws, size_t ws_size,
                              hipStream_t stream) {
  (void)in_sizes; (void)n_in; (void)out_size; (void)ws_size;
  const float* l_pos_emb = (const float*)d_in[0];
  const float* l_neg_emb = (const float*)d_in[1];
  const float* c_emb     = (const float*)d_in[2];
  const float* l_mlp_W   = (const float*)d_in[3];
  const float* l_mlp_b   = (const float*)d_in[4];
  const float* c_mlp_W   = (const float*)d_in[5];
  const float* c_mlp_b   = (const float*)d_in[6];
  const float* l_Wih     = (const float*)d_in[7];
  const float* l_Whh     = (const float*)d_in[8];
  const float* l_bih     = (const float*)d_in[9];
  const float* l_bhh     = (const float*)d_in[10];
  const float* c_Wih     = (const float*)d_in[11];
  const float* c_Whh     = (const float*)d_in[12];
  const float* c_bih     = (const float*)d_in[13];
  const float* c_bhh     = (const float*)d_in[14];
  const int*   pos_src   = (const int*)d_in[15];
  const int*   pos_dst   = (const int*)d_in[16];
  const int*   neg_src   = (const int*)d_in[17];
  const int*   neg_dst   = (const int*)d_in[18];

  char* p = (char*)d_ws;
  auto carve = [&](size_t bytes) {
    char* r = p;
    p += (bytes + 255) & ~(size_t)255;
    return r;
  };
  const size_t Lbytes = (size_t)NL_ * EMB * sizeof(float);
  const size_t Cbytes = (size_t)C_ * EMB * sizeof(float);

  float* l_hA  = (float*)carve(Lbytes);
  float* l_hB  = (float*)carve(Lbytes);
  float* l_cst = (float*)carve(Lbytes);
  float* c_h   = (float*)carve(Cbytes);
  float* c_cst = (float*)carve(Cbytes);
  float* l_msg = (float*)carve(Lbytes);
  float* c_msg = (float*)carve(Cbytes);
  float* l2c   = (float*)carve(Cbytes);
  float* c2l   = (float*)carve(Lbytes);
  _Float16* lW16  = (_Float16*)carve((size_t)3 * EMB * EMB * 2);
  _Float16* cW16  = (_Float16*)carve((size_t)3 * EMB * EMB * 2);
  _Float16* lWcat = (_Float16*)carve((size_t)384 * 512 * 2);
  _Float16* cWcat = (_Float16*)carve((size_t)256 * 512 * 2);
  float* l_bsum = (float*)carve(512 * sizeof(float));
  float* c_bsum = (float*)carve(512 * sizeof(float));

  // weight preparation (tiny): swizzle into fragment-native f16 layout
  {
    int n = EMB * EMB;  // per layer
    for (int layer = 0; layer < 3; ++layer) {
      swz_b_kernel<<<(n + 255) / 256, 256, 0, stream>>>(
          l_mlp_W + layer * n, lW16 + layer * n, EMB, EMB);
      swz_b_kernel<<<(n + 255) / 256, 256, 0, stream>>>(
          c_mlp_W + layer * n, cW16 + layer * n, EMB, EMB);
    }
    int nl = 384 * 512, nc = 256 * 512;
    build_wcat_swz_kernel<<<(nl + 255) / 256, 256, 0, stream>>>(l_Wih, l_Whh, lWcat, 256);
    build_wcat_swz_kernel<<<(nc + 255) / 256, 256, 0, stream>>>(c_Wih, c_Whh, cWcat, 128);
    add_bias_kernel<<<2, 256, 0, stream>>>(l_bih, l_bhh, l_bsum, 512);
    add_bias_kernel<<<2, 256, 0, stream>>>(c_bih, c_bhh, c_bsum, 512);
  }

  // state init
  hipMemcpyAsync(l_hA, l_pos_emb, (size_t)V_ * EMB * 4, hipMemcpyDeviceToDevice, stream);
  hipMemcpyAsync(l_hA + (size_t)V_ * EMB, l_neg_emb, (size_t)V_ * EMB * 4,
                 hipMemcpyDeviceToDevice, stream);
  hipMemcpyAsync(c_h, c_emb, Cbytes, hipMemcpyDeviceToDevice, stream);
  hipMemsetAsync(l_cst, 0, Lbytes, stream);
  hipMemsetAsync(c_cst, 0, Cbytes, stream);

  float* hcur = l_hA;
  float* hnxt = l_hB;
  const int sblocks = (E_ * 32 + 255) / 256;

  for (int r = 0; r < ROUNDS_; ++r) {
    hipMemsetAsync(l2c, 0, Cbytes, stream);
    hipMemsetAsync(c2l, 0, Lbytes, stream);

    mlp3_kernel<<<NL_ / 16, 256, 0, stream>>>(hcur, lW16, l_mlp_b, l_msg, NL_);
    mlp3_kernel<<<C_ / 16, 256, 0, stream>>>(c_h, cW16, c_mlp_b, c_msg, C_);

    // literal -> clause
    scatter_add_kernel<<<sblocks, 256, 0, stream>>>(l_msg, pos_src, pos_dst, l2c, E_, 0, 0);
    scatter_add_kernel<<<sblocks, 256, 0, stream>>>(l_msg, neg_src, neg_dst, l2c, E_, V_, 0);
    // clause -> literal (pos half / neg half of c2l)
    scatter_add_kernel<<<sblocks, 256, 0, stream>>>(c_msg, pos_dst, pos_src, c2l, E_, 0, 0);
    scatter_add_kernel<<<sblocks, 256, 0, stream>>>(c_msg, neg_dst, neg_src, c2l, E_, 0, V_);

    lstm_kernel<256><<<NL_ / 16, 256, 0, stream>>>(c2l, hcur, l_cst, hnxt, lWcat, l_bsum,
                                                   NL_, V_);
    lstm_kernel<128><<<C_ / 16, 256, 0, stream>>>(l2c, c_h, c_cst, c_h, cWcat, c_bsum,
                                                  C_, 0);
    float* tmp = hcur; hcur = hnxt; hnxt = tmp;
  }

  hipMemcpyAsync(d_out, hcur, Lbytes, hipMemcpyDeviceToDevice, stream);
  hipMemcpyAsync((float*)d_out + (size_t)NL_ * EMB, c_h, Cbytes,
                 hipMemcpyDeviceToDevice, stream);
}
